// SelfAttention_27968827031639
// MI455X (gfx1250) — matrix operand order repaired and checked by reading, not compiled, not run on hardware
//
#include <hip/hip_runtime.h>

// ---------------------------------------------------------------------------
// Self-attention for MI455X (gfx1250): fp32 I/O, bf16 WMMA math, f32 accum.
//   x:[8,2048,1024]  Wq/Wk/Wv:[1024,1024]  bq/bk/bv:[1024]  out:[8,2048,1024]
// Pipeline: cvt(fp32->bf16) -> fused QKV GEMM (V stored transposed) ->
//           per-(batch, 16-row q tile) attention with scores+softmax in LDS.
// CDNA5 paths: v_wmma_f32_16x16x32_bf16, global_load_async_to_lds_b128 (+ 
// s_wait_asynccnt), global_prefetch.
// ---------------------------------------------------------------------------

#define DHEAD 1024
#define SEQ   2048
#define NBATCH 8
#define NROWS (NBATCH * SEQ)          // 16384
#define SSTR  (SEQ + 1)               // LDS score row stride (floats)
#define QSTR  (DHEAD + 8)             // LDS Q-tile row stride (u16), bank pad
#define LDS_FLOATS (16 * SSTR + 16 * 16 + 32)
#define LDS_BYTES  ((size_t)LDS_FLOATS * 4 + (size_t)16 * QSTR * 2)

#define AS1 __attribute__((address_space(1)))
#define AS3 __attribute__((address_space(3)))

typedef __attribute__((ext_vector_type(16))) __bf16 v16bf;
typedef __attribute__((ext_vector_type(8)))  float  v8f;
typedef int v4i __attribute__((vector_size(16)));
typedef unsigned short u16;
typedef unsigned int   u32;

union FragBF { v16bf v; u32 u[8]; u16 s[16]; };

__device__ __forceinline__ u16 f32_to_bf16(float f) {
    u32 u = __float_as_uint(f);
    u = (u + 0x7FFFu + ((u >> 16) & 1u)) >> 16;   // round-to-nearest-even
    return (u16)u;
}

__device__ __forceinline__ v8f wmma_bf16(v16bf a, v16bf b, v8f c) {
    return __builtin_amdgcn_wmma_f32_16x16x32_bf16(
        /*neg_a=*/false, a, /*neg_b=*/false, b,
        /*c_mod=*/(short)0, c, /*reuse_a=*/false, /*reuse_b=*/false);
}

// A-fragment (16x32 bf16, M x K), row-major source with row stride ld (elems).
// Lane L: m = L&15, kh = L>>4. VGPR j holds K = 16*(j/4) + 8*kh + 2*(j%4) (+1).
__device__ __forceinline__ v16bf load_a_bf16(const u16* __restrict__ p, int ld) {
    const int lane = threadIdx.x & 31;
    const int m = lane & 15, kh = lane >> 4;
    FragBF fr;
    const u16* row = p + m * ld;
#pragma unroll
    for (int j = 0; j < 8; ++j) {
        int k = ((j >> 2) << 4) + (kh << 3) + ((j & 3) << 1);
        fr.u[j] = *reinterpret_cast<const u32*>(row + k);
    }
    return fr.v;
}

// B-fragment (32x16 bf16, K x N) where column n of B is row n of `p`
// (row stride ld, contiguous along K). Lane L: n = L&15, kh = L>>4.
// VGPR j holds K = 16*(j/4) + 8*kh + 2*(j%4) (+1).
__device__ __forceinline__ v16bf load_b_bf16(const u16* __restrict__ p, int ld) {
    const int lane = threadIdx.x & 31;
    const int n = lane & 15, kh = lane >> 4;
    FragBF fr;
    const u16* col = p + n * ld + (kh << 3);
#pragma unroll
    for (int j = 0; j < 8; ++j)
        fr.u[j] = *reinterpret_cast<const u32*>(col + ((j >> 2) << 4) + ((j & 3) << 1));
    return fr.v;
}

// ---------------------------------------------------------------------------
// Kernel 1: fp32 -> bf16 conversion (4 elems / thread; n multiple of 1024)
// ---------------------------------------------------------------------------
__global__ __launch_bounds__(256) void cvt_bf16_kernel(
    const float* __restrict__ src, u16* __restrict__ dst, int n)
{
    int i = (blockIdx.x * 256 + threadIdx.x) * 4;
    if (i + 3 >= n) return;
    float4 f = *reinterpret_cast<const float4*>(src + i);
    u32 lo = (u32)f32_to_bf16(f.x) | ((u32)f32_to_bf16(f.y) << 16);
    u32 hi = (u32)f32_to_bf16(f.z) | ((u32)f32_to_bf16(f.w) << 16);
    *reinterpret_cast<u32*>(dst + i)     = lo;
    *reinterpret_cast<u32*>(dst + i + 2) = hi;
}

// ---------------------------------------------------------------------------
// Kernel 2: fused QKV projection.  Q = x*Wq^T + bq  (same for K, V).
// Grid: (NROWS/64, DHEAD/128). Block 256 = 8 waves; wave (wr,wc) does a
// 16x64 tile for all three outputs, reusing the x A-fragment 12x per K-step.
// V is written transposed per batch: Vt[(b*1024 + e)*2048 + s].
// ---------------------------------------------------------------------------
__global__ __launch_bounds__(256) void qkv_kernel(
    const u16* __restrict__ xb,
    const u16* __restrict__ Wqb, const u16* __restrict__ Wkb, const u16* __restrict__ Wvb,
    const float* __restrict__ bq, const float* __restrict__ bk, const float* __restrict__ bv,
    u16* __restrict__ Qb, u16* __restrict__ Kb, u16* __restrict__ Vt)
{
    const int wave = threadIdx.x >> 5;
    const int lane = threadIdx.x & 31;
    const int wr = wave >> 1;                       // 0..3
    const int wc = wave & 1;                        // 0..1
    const int row0 = blockIdx.x * 64 + wr * 16;     // query-row tile
    const int col0 = blockIdx.y * 128 + wc * 64;    // output-feature tile
    const int n = lane & 15, h = lane >> 4;

    v8f acc[3][4] = {};

    for (int d0 = 0; d0 < DHEAD; d0 += 32) {
        v16bf a = load_a_bf16(xb + row0 * DHEAD + d0, DHEAD);
        const u16* Ws[3] = { Wqb, Wkb, Wvb };
#pragma unroll
        for (int wi = 0; wi < 3; ++wi) {
#pragma unroll
            for (int ct = 0; ct < 4; ++ct) {
                v16bf b = load_b_bf16(Ws[wi] + (col0 + ct * 16) * DHEAD + d0, DHEAD);
                acc[wi][ct] = wmma_bf16(a, b, acc[wi][ct]);
            }
        }
    }

    // Epilogue: bias + bf16 stores (C layout: row = r + 8*h, col = n)
#pragma unroll
    for (int ct = 0; ct < 4; ++ct) {
        const int col = col0 + ct * 16 + n;
        const float vbq = bq[col], vbk = bk[col], vbv = bv[col];
#pragma unroll
        for (int r = 0; r < 8; ++r) {
            const int row = row0 + r + 8 * h;
            Qb[row * DHEAD + col] = f32_to_bf16(acc[0][ct][r] + vbq);
            Kb[row * DHEAD + col] = f32_to_bf16(acc[1][ct][r] + vbk);
            const int bidx = row >> 11;             // row / SEQ
            const int s    = row & (SEQ - 1);
            Vt[((bidx << 10) + col) * SEQ + s] = f32_to_bf16(acc[2][ct][r] + vbv);
        }
    }
}

// ---------------------------------------------------------------------------
// Kernel 3: attention for one (batch, 16-query-row) tile.
// Phase 0: async-stage the shared 16x1024 Q tile into padded LDS
// Phase 1: scores = scale * Q Kt (16 x 2048 fp32 in LDS); each wave owns 16
//          contiguous key tiles, 2 blocks of 8 accumulators -> one Q
//          A-fragment feeds 8 WMMAs.
// Phase 2: row softmax (two-level LDS reduction); 1/sum deferred to epilogue
// Phase 3: O = P V using Vt (contiguous-K B-fragments)
// Block 256 = 8 waves. LDS: sc[16][2049] f32 + red[256] + rowstat[32] +
//                           qtile[16][QSTR] u16.
// ---------------------------------------------------------------------------
__global__ __launch_bounds__(256) void attn_kernel(
    const u16* __restrict__ Qb, const u16* __restrict__ Kb,
    const u16* __restrict__ Vt, float* __restrict__ out)
{
    extern __shared__ float lds[];
    float* sc      = lds;                       // 16 * SSTR
    float* red     = lds + 16 * SSTR;           // 16 * 16
    float* rowstat = red + 256;                 // [0..15]=rowmax, [16..31]=1/sum
    u16*   qtile   = (u16*)(lds + LDS_FLOATS);  // 16 * QSTR bf16

    const int b    = blockIdx.y;
    const int q0   = blockIdx.x * 16;
    const int wave = threadIdx.x >> 5;
    const int lane = threadIdx.x & 31;
    const int n = lane & 15, h = lane >> 4;

    const u16* Qbase = Qb + (b * SEQ + q0) * DHEAD;
    const u16* Kbase = Kb + (b * SEQ) * DHEAD;
    const u16* Vbase = Vt + b * (DHEAD * SEQ);  // [e][s]
    const float scale = 0.03125f;               // 1/sqrt(1024)

    // ---- Phase 0: stage Q tile (16 x 1024 bf16 = 32 KB) into padded LDS ----
    {
        // 2048 x 16-byte chunks; 8 per thread. Row stride pad keeps fragment
        // reads conflict-free (dword stride 516 -> lane banks 4m+c, all 64).
        for (int i = threadIdx.x; i < 2048; i += 256) {
            const int row = i >> 7, c = i & 127;          // 128 chunks / row
            const u16* src = Qbase + row * DHEAD + c * 8;
            u16*       dst = qtile + row * QSTR  + c * 8;
#if __has_builtin(__builtin_amdgcn_global_load_async_to_lds_b128)
            // Signature (probe-derived): (int4 AS1*, int4 AS3*, i32 off, i32 cpol)
            __builtin_amdgcn_global_load_async_to_lds_b128(
                (AS1 v4i*)(uintptr_t)src,
                (AS3 v4i*)(u32)(uintptr_t)dst, 0, 0);
#else
            // Inline-asm async copy: vdst = LDS byte address, vaddr = 64-bit
            // global address (saddr disabled).
            u32 ldsaddr = (u32)(uintptr_t)dst;
            unsigned long long gaddr = (unsigned long long)(uintptr_t)src;
            asm volatile("global_load_async_to_lds_b128 %0, %1, off"
                         :: "v"(ldsaddr), "v"(gaddr) : "memory");
#endif
        }
#if __has_builtin(__builtin_amdgcn_s_wait_asynccnt)
        __builtin_amdgcn_s_wait_asynccnt(0);
#else
        asm volatile("s_wait_asynccnt 0x0" ::: "memory");
#endif
    }
    __syncthreads();

    // ---- Phase 1: scores into LDS (wave owns key tiles [wave*16, wave*16+16)) ----
#pragma unroll
    for (int ktb = 0; ktb < 2; ++ktb) {
        const int ktbase = wave * 16 + ktb * 8;
        __builtin_prefetch(Kbase + (ktbase + 8) * 16 * DHEAD, 0, 1);
        v8f acc[8] = {};
        for (int d0 = 0; d0 < DHEAD; d0 += 32) {
            v16bf a = load_a_bf16(qtile + d0, QSTR);          // LDS, reused 8x
#pragma unroll
            for (int i = 0; i < 8; ++i) {
                v16bf bb = load_b_bf16(Kbase + (ktbase + i) * 16 * DHEAD + d0, DHEAD);
                acc[i] = wmma_bf16(a, bb, acc[i]);
            }
        }
#pragma unroll
        for (int i = 0; i < 8; ++i)
#pragma unroll
            for (int r = 0; r < 8; ++r)
                sc[(r + 8 * h) * SSTR + (ktbase + i) * 16 + n] = acc[i][r] * scale;
    }
    __syncthreads();

    // ---- Phase 2: softmax over each of the 16 rows (2048 wide) ----
    {
        const int row = threadIdx.x & 15;
        const int c   = threadIdx.x >> 4;       // 16 chunks of 128
        float* rp = sc + row * SSTR + c * 128;

        float lmax = -3.4e38f;
        for (int i = 0; i < 128; ++i) lmax = fmaxf(lmax, rp[i]);
        red[row * 16 + c] = lmax;
        __syncthreads();
        if (c == 0) {
            float mx = red[row * 16];
            for (int i = 1; i < 16; ++i) mx = fmaxf(mx, red[row * 16 + i]);
            rowstat[row] = mx;
        }
        __syncthreads();
        const float mx = rowstat[row];
        float lsum = 0.f;
        for (int i = 0; i < 128; ++i) {
            float e = __expf(rp[i] - mx);
            rp[i] = e;
            lsum += e;
        }
        red[row * 16 + c] = lsum;
        __syncthreads();
        if (c == 0) {
            float sum = 0.f;
            for (int i = 0; i < 16; ++i) sum += red[row * 16 + i];
            rowstat[16 + row] = 1.0f / sum;
        }
        __syncthreads();
    }

    // ---- Phase 3: O = P * V ; each wave owns 128 output features ----
    const int col0 = wave * 128;
    v8f acc[8] = {};
    for (int k0 = 0; k0 < SEQ; k0 += 32) {
        // Build A-fragment from LDS fp32 probabilities (convert to bf16)
        FragBF a;
        {
            const int m = lane & 15, kh = lane >> 4;
            const float* prow = sc + m * SSTR + k0;
#pragma unroll
            for (int j = 0; j < 8; ++j) {
                int k = ((j >> 2) << 4) + (kh << 3) + ((j & 3) << 1);
                a.u[j] = (u32)f32_to_bf16(prow[k]) |
                         ((u32)f32_to_bf16(prow[k + 1]) << 16);
            }
        }
#pragma unroll
        for (int ct = 0; ct < 8; ++ct) {
            v16bf bb = load_b_bf16(Vbase + (col0 + ct * 16) * SEQ + k0, SEQ);
            acc[ct] = wmma_bf16(a.v, bb, acc[ct]);
        }
    }

    float* obase = out + (b * SEQ + q0) * DHEAD;
#pragma unroll
    for (int ct = 0; ct < 8; ++ct) {
#pragma unroll
        for (int r = 0; r < 8; ++r) {
            const int mrow = r + 8 * h;
            obase[mrow * DHEAD + col0 + ct * 16 + n] =
                acc[ct][r] * rowstat[16 + mrow];
        }
    }
}

// ---------------------------------------------------------------------------
// Host-side launcher
// ---------------------------------------------------------------------------
extern "C" void kernel_launch(void* const* d_in, const int* in_sizes, int n_in,
                              void* d_out, int out_size, void* d_ws, size_t ws_size,
                              hipStream_t stream)
{
    const float* x  = (const float*)d_in[0];
    const float* Wq = (const float*)d_in[1];
    const float* bq = (const float*)d_in[2];
    const float* Wk = (const float*)d_in[3];
    const float* bk = (const float*)d_in[4];
    const float* Wv = (const float*)d_in[5];
    const float* bv = (const float*)d_in[6];
    float* out = (float*)d_out;

    // Workspace layout (ushort elements): xb | Wqb | Wkb | Wvb | Qb | Kb | Vt
    const size_t NX = (size_t)NROWS * DHEAD;        // 16,777,216
    const size_t NW = (size_t)DHEAD * DHEAD;        //  1,048,576
    u16* xb  = (u16*)d_ws;
    u16* Wqb = xb  + NX;
    u16* Wkb = Wqb + NW;
    u16* Wvb = Wkb + NW;
    u16* Qb  = Wvb + NW;
    u16* Kb  = Qb  + NX;
    u16* Vt  = Kb  + NX;                            // total ~134 MB

    // 1) fp32 -> bf16 conversions
    cvt_bf16_kernel<<<(int)(NX / 1024), 256, 0, stream>>>(x,  xb,  (int)NX);
    cvt_bf16_kernel<<<(int)(NW / 1024), 256, 0, stream>>>(Wq, Wqb, (int)NW);
    cvt_bf16_kernel<<<(int)(NW / 1024), 256, 0, stream>>>(Wk, Wkb, (int)NW);
    cvt_bf16_kernel<<<(int)(NW / 1024), 256, 0, stream>>>(Wv, Wvb, (int)NW);

    // 2) fused QKV projection
    qkv_kernel<<<dim3(NROWS / 64, DHEAD / 128), 256, 0, stream>>>(
        xb, Wqb, Wkb, Wvb, bq, bk, bv, Qb, Kb, Vt);

    // 3) attention (scores + softmax in LDS, then P*V)
    attn_kernel<<<dim3(SEQ / 16, NBATCH), 256, LDS_BYTES, stream>>>(
        Qb, Kb, Vt, out);
}
